// AdaptiveMixing_56573309224538
// MI455X (gfx1250) — matrix operand order repaired and checked by reading, not compiled, not run on hardware
//
#include <hip/hip_runtime.h>
#include <hip/hip_bf16.h>
#include <math.h>

// ---------------------------------------------------------------------------
// AdaptiveMixing fused kernel for MI455X (gfx1250, wave32, WMMA).
//
// v2: Wp/Wo are pre-converted to f16 and pre-swizzled into WMMA A-operand
// register order in d_ws (33.5MB), so the two dominant GEMMs load A with
// 2x global_load_b128 per wmma (zero VALU) and B with 2x ds_load_b128
// (the dense B lane layout makes each lane's 16 f16 two contiguous 8-runs in LDS).
// ---------------------------------------------------------------------------

typedef _Float16 f16;
typedef _Float16 v16h __attribute__((ext_vector_type(16)));
typedef float    v8f  __attribute__((ext_vector_type(8)));

#define QT       16     // queries per workgroup
#define QDIM     256    // query / output channel dim
#define NGRP     4
#define TOT      8192   // params per (token, group)
#define HALFTOT  4096   // M region rows == S region rows
#define OUTP     128
#define INP      32
#define EFF      64

#define SQ_ELEMS (QT * QDIM)     //  4096 f16 ->   8KB
#define SP_ELEMS (QT * HALFTOT)  // 65536 f16 -> 128KB
#define SF_ELEMS (QT * 1024)     // 16384 f16 ->  32KB
#define SMEM_BYTES ((SQ_ELEMS + SP_ELEMS + SF_ELEMS) * 2)

#define WP_ELEMS (32768 * 256)   // swizzled f16 Wp in ws
#define WO_ELEMS (256 * 32768)   // swizzled f16 Wo in ws

__device__ __forceinline__ v8f wmma_f16(v16h a, v16h b, v8f c) {
  return __builtin_amdgcn_wmma_f32_16x16x32_f16(false, a, false, b, (short)0, c,
                                                false, false);
}

__device__ __forceinline__ v8f zero8() {
  v8f z = {0.f, 0.f, 0.f, 0.f, 0.f, 0.f, 0.f, 0.f};
  return z;
}

// --- A operand (16x32 f16): per lane, two contiguous K-runs of 8:
//     k in [half*8, half*8+8) and [16+half*8, 16+half*8+8).
// Pre-swizzled weights store those 16 f16 contiguously per (tile, lane):
// one 32B slot -> 2x global_load_b128, zero VALU.
__device__ __forceinline__ v16h load_a_ws(const f16* __restrict__ tilep,
                                          int lane) {
  return *(const v16h*)(tilep + lane * 16);
}

// A operand from row-major f32 (used for x in mix-1 and the swizzle pre-pass): float4 runs + cvt.
__device__ __forceinline__ v16h load_a_f32(const float* __restrict__ rowp,
                                           int half) {
  const float4 r0 = *(const float4*)(rowp + half * 8);
  const float4 r1 = *(const float4*)(rowp + half * 8 + 4);
  const float4 r2 = *(const float4*)(rowp + 16 + half * 8);
  const float4 r3 = *(const float4*)(rowp + 16 + half * 8 + 4);
  v16h a;
  a[0] = (f16)r0.x;  a[1] = (f16)r0.y;  a[2] = (f16)r0.z;  a[3] = (f16)r0.w;
  a[4] = (f16)r1.x;  a[5] = (f16)r1.y;  a[6] = (f16)r1.z;  a[7] = (f16)r1.w;
  a[8] = (f16)r2.x;  a[9] = (f16)r2.y;  a[10] = (f16)r2.z; a[11] = (f16)r2.w;
  a[12] = (f16)r3.x; a[13] = (f16)r3.y; a[14] = (f16)r3.z; a[15] = (f16)r3.w;
  return a;
}

// --- B operand (32x16 f16): lane's 16 f16 are k in [half*8, half*8+8) and
// [16+half*8, 16+half*8+8), matching A -> 2x ds_load_b128.
__device__ __forceinline__ v16h load_b_lds(const f16* colp, int half) {
  typedef _Float16 v8h_ __attribute__((ext_vector_type(8)));
  const v8h_ lo = *(const v8h_*)(colp + half * 8);
  const v8h_ hi = *(const v8h_*)(colp + 16 + half * 8);
  return __builtin_shufflevector(lo, hi, 0, 1, 2, 3, 4, 5, 6, 7, 8, 9, 10, 11, 12, 13, 14, 15);
}

__device__ __forceinline__ float wave_sum(float v) {
#pragma unroll
  for (int m = 1; m < 32; m <<= 1) v += __shfl_xor(v, m, 32);
  return v;
}

__device__ __forceinline__ float gelu_exact(float v) {
  return 0.5f * v * (1.0f + erff(v * 0.70710678118654752440f));
}

// ---------------------------------------------------------------------------
// Pre-pass: convert a row-major f32 matrix [rows x K] into f16 swizzled into
// WMMA A-operand order: tile t = rt*(K/32)+kt holds 32 lanes x 16 f16,
// lane slot = the lane's two contiguous K-runs of 8. One thread per lane-slot.
// ---------------------------------------------------------------------------
__global__ __launch_bounds__(256) void swizzleA_f32_to_f16(
    const float* __restrict__ src, f16* __restrict__ dst, int rows, int K) {
  const int t = blockIdx.x * 256 + threadIdx.x;
  const int lane = t & 31;
  const int tile = t >> 5;
  const int nkt = K >> 5;
  const int rt = tile / nkt, kt = tile - rt * nkt;
  if (rt * 16 >= rows) return;
  const int l16 = lane & 15, half = lane >> 4;
  const float* rowp = src + (size_t)(rt * 16 + l16) * K + kt * 32;
  v16h a = load_a_f32(rowp, half);
  *(v16h*)(dst + ((size_t)tile * 32 + lane) * 16) = a;
}

// ---------------------------------------------------------------------------
// Cooperative param generation: 4096 rows x 16 queries into sParams (f16).
// ph==0: M region stored transposed (M^T: [d*64+c]); ph==1: S natural [o*32+p].
// ---------------------------------------------------------------------------
__device__ __forceinline__ void gen_params(const f16* __restrict__ wsWp,
                                           const float* __restrict__ bp,
                                           const f16* sQ, f16* sParams, int g,
                                           int ph, int wave, int lane, int l16,
                                           int half) {
  const float* bpp = bp + g * TOT + ph * HALFTOT;
  f16* dst = sParams + l16 * HALFTOT;  // D-tile column n = lane%16 = query
  for (int t = 0; t < 16; ++t) {       // 16 row-tiles per wave
    const int rbase = wave * 256 + t * 16;
    const int rt = g * 512 + ph * 256 + (rbase >> 4);  // Wp row-tile index
    const f16* atile = wsWp + (size_t)rt * 8 * 32 * 16;  // 8 K-tiles / row-tile
    const f16* qcol = sQ + l16 * QDIM;
    v8f acc = zero8();
    for (int k0 = 0; k0 < QDIM; k0 += 32) {
      const f16* tp = atile + (size_t)(k0 >> 5) * 32 * 16;
      if (k0 + 32 < QDIM) __builtin_prefetch(tp + 32 * 16, 0, 0);
      v16h a = load_a_ws(tp, lane);
      v16h b = load_b_lds(qcol + k0, half);
      acc = wmma_f16(a, b, acc);
    }
#pragma unroll
    for (int r = 0; r < 8; ++r) {  // C layout: row = rbase + r + 8*half
      const int row = rbase + r + 8 * half;
      const float val = acc[r] + bpp[row];
      const int sidx = ph ? row : ((row & 63) * 64 + (row >> 6));  // transpose M
      dst[sidx] = (f16)val;
    }
  }
}

__global__ __launch_bounds__(512) void AdaptiveMixing_fused_kernel(
    const float* __restrict__ x, const float* __restrict__ query,
    const f16* __restrict__ wsWp, const float* __restrict__ bp,
    const f16* __restrict__ wsWo, const float* __restrict__ bo,
    float* __restrict__ out) {
  extern __shared__ char smem_raw[];
  f16* sQ      = (f16*)smem_raw;            // [q][k]       4096 f16
  f16* sParams = sQ + SQ_ELEMS;             // [q][row]    65536 f16
  f16* sFeat   = sParams + SP_ELEMS;        // [q][fLocal] 16384 f16

  const int tid  = threadIdx.x;
  const int lane = tid & 31;
  const int wave = tid >> 5;      // 0..15; this wave owns query `wave`
  const int half = lane >> 4;
  const int l16  = lane & 15;
  const int bn0  = blockIdx.x * QT;
  const int myq  = wave;
  const int mybn = bn0 + myq;

  // ---- stage the query tile into LDS as f16 (B operand for GEMM1) ----
  for (int i = tid; i < SQ_ELEMS; i += 512) {
    const int q = i >> 8, k = i & 255;
    sQ[q * QDIM + k] = (f16)query[(size_t)(bn0 + q) * QDIM + k];
  }
  __syncthreads();

  v8f C3 = zero8();  // projection acc: rows qd = 16*wave..+15, cols = 16 queries

  for (int g = 0; g < NGRP; ++g) {
    // ================= GEMM1/M: generate channel-mix params ================
    gen_params(wsWp, bp, sQ, sParams, g, /*ph=*/0, wave, lane, l16, half);
    __syncthreads();

    // ================= mix-1: out1(32x64) = x(32x64) @ M(64x64) ============
    v8f o1[2][4];
#pragma unroll
    for (int pc = 0; pc < 2; ++pc)
#pragma unroll
      for (int dc = 0; dc < 4; ++dc) o1[pc][dc] = zero8();

    const float* xg = x + ((size_t)mybn * NGRP + g) * (INP * EFF);
    for (int kc = 0; kc < 2; ++kc) {  // contraction c in chunks of 32
      v16h ax[2];
#pragma unroll
      for (int pc = 0; pc < 2; ++pc)
        ax[pc] = load_a_f32(xg + (pc * 16 + l16) * EFF + kc * 32, half);
#pragma unroll
      for (int dc = 0; dc < 4; ++dc) {
        v16h bm = load_b_lds(
            sParams + myq * HALFTOT + (dc * 16 + l16) * 64 + kc * 32, half);
#pragma unroll
        for (int pc = 0; pc < 2; ++pc) o1[pc][dc] = wmma_f16(ax[pc], bm, o1[pc][dc]);
      }
    }

    // ---- LayerNorm over (32,64) + exact GELU, all in registers ----
    float s = 0.f, ss = 0.f;
#pragma unroll
    for (int pc = 0; pc < 2; ++pc)
#pragma unroll
      for (int dc = 0; dc < 4; ++dc)
#pragma unroll
        for (int r = 0; r < 8; ++r) {
          const float v = o1[pc][dc][r];
          s += v; ss += v * v;
        }
    s = wave_sum(s); ss = wave_sum(ss);
    const float mean1 = s * (1.f / 2048.f);
    const float inv1  = rsqrtf(ss * (1.f / 2048.f) - mean1 * mean1 + 1e-5f);

    // ---- pack A2 = out1^T (A operand of mix-2) purely in-lane:
    // C layout gives lane = d, p striped over VGPRs exactly matching the
    // A-operand half-lane K split -> no LDS round-trip for the chained GEMM.
    v16h A2[4];
#pragma unroll
    for (int dc = 0; dc < 4; ++dc)
#pragma unroll
      for (int v = 0; v < 4; ++v) {
        A2[dc][2 * v]     = (f16)gelu_exact((o1[0][dc][2 * v]     - mean1) * inv1);
        A2[dc][2 * v + 1] = (f16)gelu_exact((o1[0][dc][2 * v + 1] - mean1) * inv1);
        A2[dc][8 + 2 * v] = (f16)gelu_exact((o1[1][dc][2 * v]     - mean1) * inv1);
        A2[dc][9 + 2 * v] = (f16)gelu_exact((o1[1][dc][2 * v + 1] - mean1) * inv1);
      }
    __syncthreads();  // everyone done reading the M region

    // ================= GEMM1/S: generate spatial-mix params ================
    gen_params(wsWp, bp, sQ, sParams, g, /*ph=*/1, wave, lane, l16, half);
    __syncthreads();

    // ======== mix-2 pass 1 (stats only): out2^T = out1^T @ S^T =============
    float s2 = 0.f, ss2 = 0.f;
    for (int oc = 0; oc < 8; ++oc) {
      v16h bs = load_b_lds(sParams + myq * HALFTOT + (oc * 16 + l16) * INP, half);
#pragma unroll
      for (int dc = 0; dc < 4; ++dc) {
        v8f c2 = wmma_f16(A2[dc], bs, zero8());
#pragma unroll
        for (int r = 0; r < 8; ++r) {
          const float v = c2[r];
          s2 += v; ss2 += v * v;
        }
      }
    }
    s2 = wave_sum(s2); ss2 = wave_sum(ss2);
    const float mean2 = s2 * (1.f / 8192.f);
    const float inv2  = rsqrtf(ss2 * (1.f / 8192.f) - mean2 * mean2 + 1e-5f);

    // ==== mix-2 pass 2 (recompute, cheap) + projection GEMM ================
    for (int oc = 0; oc < 8; ++oc) {
      v16h bs = load_b_lds(sParams + myq * HALFTOT + (oc * 16 + l16) * INP, half);
#pragma unroll
      for (int dc = 0; dc < 4; ++dc) {
        v8f c2 = wmma_f16(A2[dc], bs, zero8());
#pragma unroll
        for (int r = 0; r < 8; ++r) {
          const float v = gelu_exact((c2[r] - mean2) * inv2);
          // fLocal = (o - oc*16)*64 + d; lane n = o, row m = d
          sFeat[myq * 1024 + l16 * 64 + dc * 16 + 8 * half + r] = (f16)v;
        }
      }
      __syncthreads();  // feature chunk complete for all 16 queries

      // GEMM3 K-chunk: wave accumulates its 16 output rows vs 16 queries.
      // Wo row-tile rt = wave; K-tile index = (g*8192 + oc*1024 + k0)/32.
      const f16* wo_tiles =
          wsWo + (((size_t)wave * 1024 + (size_t)(g * 8192 + oc * 1024) / 32) *
                  32 * 16);
      const f16* featcol = sFeat + l16 * 1024;  // B column n = query
      for (int k0 = 0; k0 < 1024; k0 += 32) {
        const f16* tp = wo_tiles + (size_t)(k0 >> 5) * 32 * 16;
        if (k0 + 32 < 1024) __builtin_prefetch(tp + 32 * 16, 0, 0);
        v16h aw = load_a_ws(tp, lane);
        v16h bf = load_b_lds(featcol + k0, half);
        C3 = wmma_f16(aw, bf, C3);
      }
      __syncthreads();  // before next oc overwrites sFeat / next g sParams
    }
  }

  // ================= epilogue: bias + residual + ReLU ======================
  {
    const int bno = bn0 + l16;  // C3 column n = query
#pragma unroll
    for (int r = 0; r < 8; ++r) {
      const int qd = 16 * wave + 8 * half + r;
      const float v = C3[r] + bo[qd] + query[(size_t)bno * QDIM + qd];
      out[(size_t)bno * QDIM + qd] = fmaxf(v, 0.f);
    }
  }
}

extern "C" void kernel_launch(void* const* d_in, const int* in_sizes, int n_in,
                              void* d_out, int out_size, void* d_ws,
                              size_t ws_size, hipStream_t stream) {
  const float* x     = (const float*)d_in[0];
  const float* query = (const float*)d_in[1];
  const float* Wp    = (const float*)d_in[2];
  const float* bp    = (const float*)d_in[3];
  const float* Wo    = (const float*)d_in[4];
  const float* bo    = (const float*)d_in[5];
  float* out = (float*)d_out;

  f16* wsWp = (f16*)d_ws;            // 16.8 MB swizzled f16 Wp
  f16* wsWo = wsWp + WP_ELEMS;       // 16.8 MB swizzled f16 Wo

  // --- pre-pass: f32 -> f16 + WMMA A-operand swizzle (runs per launch) ---
  {
    const int tilesWp = (32768 / 16) * (256 / 32);   // 16384
    const int tilesWo = (256 / 16) * (32768 / 32);   // 16384
    swizzleA_f32_to_f16<<<tilesWp * 32 / 256, 256, 0, stream>>>(Wp, wsWp,
                                                                32768, 256);
    swizzleA_f32_to_f16<<<tilesWo * 32 / 256, 256, 0, stream>>>(Wo, wsWo, 256,
                                                                32768);
  }

  const int bnTotal = in_sizes[1] / QDIM;  // B*N = 3600
  const int nwg = bnTotal / QT;            // 225 workgroups

  // 168KB dynamic LDS (CDNA5 allows up to 320KB per workgroup).
  (void)hipFuncSetAttribute((const void*)AdaptiveMixing_fused_kernel,
                            hipFuncAttributeMaxDynamicSharedMemorySize,
                            SMEM_BYTES);

  AdaptiveMixing_fused_kernel<<<nwg, 512, SMEM_BYTES, stream>>>(
      x, query, wsWp, bp, wsWo, bo, out);
}
